// LSTMCell_40733469835813
// MI455X (gfx1250) — matrix operand order repaired and checked by reading, not compiled, not run on hardware
//
#include <hip/hip_runtime.h>

// ---------------------------------------------------------------------------
// LSTM on MI455X (gfx1250, wave32, WMMA).
//
//  - Persistent kernel: 64 WGs x 256 threads; WG owns 16 hidden columns.
//  - Weight slice (2048 K x 64 gate-cols, bf16 = 256 KB) pinned in LDS,
//    staged with GLOBAL_LOAD_ASYNC_TO_LDS_B128 (ASYNCcnt) when available.
//  - Gates via v_wmma_f32_16x16x32_bf16; A-operands (x_t, a(t-1)) stream
//    from L2 as global_load_b128; B-fragments are 32B ds_load reads.
//  - a(t) bf16 double-buffered in workspace; one grid barrier per step.
// ---------------------------------------------------------------------------

#define NWG     64
#define BLK     256
#define BATCH   64
#define TSTEPS  512
#define DIN     1024
#define DH      1024
#define KTOT    2048          // DIN + DH
#define HC      16            // hidden columns per workgroup
#define SLICE_COLS 64         // 4 gates * HC

#define WB_ELEMS  ((size_t)NWG * SLICE_COLS * KTOT)       // packed weights (bf16)
#define XB_ELEMS  ((size_t)TSTEPS * BATCH * DIN)          // packed x (bf16)
#define AB_ELEMS  ((size_t)2 * BATCH * DH)                // double-buffered a (bf16)

#define SMEM_W_BYTES (SLICE_COLS * KTOT * 2)              // 262144
#define SMEM_G_BYTES (BATCH * SLICE_COLS * 4)             // 16384 gate staging
#define SMEM_BYTES   (SMEM_W_BYTES + SMEM_G_BYTES)        // 278528 <= 320KB

typedef __attribute__((ext_vector_type(16))) __bf16 v16bf;
typedef __attribute__((ext_vector_type(8)))  float  v8f;
typedef __attribute__((ext_vector_type(4)))  int    vint4;   // trivial 128-bit

#define AS1 __attribute__((address_space(1)))
#define AS3 __attribute__((address_space(3)))

#if defined(__has_builtin)
#  if __has_builtin(__builtin_amdgcn_global_load_async_to_lds_b128) && \
      __has_builtin(__builtin_amdgcn_s_wait_asynccnt)
#    define HAVE_ASYNC_LDS 1
#  endif
#endif

union Frag16 {
  v16bf v;
  vint4 q[2];
};

__device__ __forceinline__ unsigned short f2bf(float f) {
  union { float f; unsigned u; } x; x.f = f;
  unsigned r = x.u + 0x7FFFu + ((x.u >> 16) & 1u);   // round-to-nearest-even
  return (unsigned short)(r >> 16);
}

__device__ __forceinline__ float sigmoidf_(float x) {
  return 1.0f / (1.0f + __expf(-x));
}

// ---------------------------------------------------------------------------
// Pack weights: slice[wg][c][k] = W_gate[k][wg*16 + c%16], gate = c/16.
// ---------------------------------------------------------------------------
__global__ void k_pack_w(const float* __restrict__ Wf, const float* __restrict__ Wi,
                         const float* __restrict__ Wo, const float* __restrict__ Wc,
                         unsigned short* __restrict__ Wb) {
  const int g = blockIdx.x;
  unsigned short* dst = Wb + (size_t)g * SLICE_COLS * KTOT;
  for (int idx = threadIdx.x; idx < SLICE_COLS * KTOT; idx += BLK) {
    const int c    = idx / KTOT;
    const int k    = idx - c * KTOT;
    const int gate = c >> 4;
    const int col  = g * HC + (c & 15);
    const float* Wsrc = (gate == 0) ? Wf : (gate == 1) ? Wi : (gate == 2) ? Wo : Wc;
    dst[idx] = f2bf(Wsrc[(size_t)k * DH + col]);
  }
}

// ---------------------------------------------------------------------------
// Pack x: (B,T,D) fp32 -> (T,B,D) bf16, 8 elements per thread along D.
// ---------------------------------------------------------------------------
__global__ void k_pack_x(const float* __restrict__ x, unsigned short* __restrict__ xb) {
  const size_t base = ((size_t)blockIdx.x * BLK + threadIdx.x) * 8;
  if (base >= (size_t)BATCH * TSTEPS * DIN) return;
  const size_t d  = base % DIN;
  const size_t bt = base / DIN;
  const size_t t  = bt % TSTEPS;
  const size_t b  = bt / TSTEPS;
  const float* src = x + base;
  unsigned short* dst = xb + ((t * BATCH + b) * DIN + d);
#pragma unroll
  for (int j = 0; j < 8; ++j) dst[j] = f2bf(src[j]);
}

// ---------------------------------------------------------------------------
// Zero a-buffers and the grid-barrier counter (every launch: deterministic).
// ---------------------------------------------------------------------------
__global__ void k_init(unsigned short* __restrict__ ab, unsigned* __restrict__ bar) {
  for (int i = threadIdx.x; i < (int)AB_ELEMS; i += BLK) ab[i] = 0;
  if (threadIdx.x == 0) *bar = 0u;
}

// ---------------------------------------------------------------------------
// One K=32 slab: two row-tiles of A against the shared B fragment.
// ---------------------------------------------------------------------------
__device__ __forceinline__ void mma_k32(const unsigned short* __restrict__ As,
                                        int k0,
                                        const unsigned short* __restrict__ bptr,
                                        int r0, int am, int ak,
                                        v8f& acc0, v8f& acc1) {
  Frag16 bfrag, af0, af1;
  bfrag.q[0] = *(const vint4*)(bptr);
  bfrag.q[1] = *(const vint4*)(bptr + 16);

  const unsigned short* a0p = As + (size_t)(r0 * 16 + am) * 1024 + k0 + ak;
  const unsigned short* a1p = a0p + 16 * 1024;
  af0.q[0] = *(const vint4*)(a0p);
  af0.q[1] = *(const vint4*)(a0p + 16);
  af1.q[0] = *(const vint4*)(a1p);
  af1.q[1] = *(const vint4*)(a1p + 16);

  acc0 = __builtin_amdgcn_wmma_f32_16x16x32_bf16(false, af0.v, false, bfrag.v,
                                                 (short)0, acc0, false, false);
  acc1 = __builtin_amdgcn_wmma_f32_16x16x32_bf16(false, af1.v, false, bfrag.v,
                                                 (short)0, acc1, false, false);
}

// ---------------------------------------------------------------------------
// Persistent recurrent kernel: 64 WGs x 256 threads (8 wave32).
// Wave w: gate q = w>>1, batch row-tiles r0=(w&1)*2, r0+1.
// ---------------------------------------------------------------------------
__launch_bounds__(BLK, 1)
__global__ void k_lstm(const unsigned short* __restrict__ xb,
                       const unsigned short* __restrict__ Wb,
                       unsigned short* __restrict__ ab,
                       const float* __restrict__ bF, const float* __restrict__ bI,
                       const float* __restrict__ bO, const float* __restrict__ bC,
                       float* __restrict__ out, unsigned* __restrict__ bar) {
  extern __shared__ char smem[];
  unsigned short* sW = (unsigned short*)smem;
  float*          sG = (float*)(smem + SMEM_W_BYTES);

  const int wg   = blockIdx.x;
  const int tid  = threadIdx.x;
  const int wave = tid >> 5;
  const int lane = tid & 31;

  // ---- stage weight slice (256 KB) into LDS once ----
#if defined(HAVE_ASYNC_LDS)
  {
    // per-lane async DMA: 16 bytes global -> LDS, no VGPR round-trip
    const char* gsrc  = (const char*)(Wb + (size_t)wg * SLICE_COLS * KTOT);
    char*       lbase = (char*)sW;
    const int n16 = SLICE_COLS * KTOT / 8;         // 16384 16-byte chunks
    for (int i = tid; i < n16; i += BLK) {
      AS1 vint4* g = (AS1 vint4*)(gsrc + (size_t)i * 16);
      AS3 vint4* l = (AS3 vint4*)(lbase + (size_t)i * 16);
      __builtin_amdgcn_global_load_async_to_lds_b128(g, l, 0, 0);
    }
    __builtin_amdgcn_s_wait_asynccnt(0);
  }
#else
  {
    const vint4* src = (const vint4*)(Wb + (size_t)wg * SLICE_COLS * KTOT);
    vint4*       dst = (vint4*)sW;
    const int n16 = SLICE_COLS * KTOT / 8;
    for (int i = tid; i < n16; i += BLK) dst[i] = src[i];
  }
#endif
  __syncthreads();

  const int q    = wave >> 1;
  const int r0   = (wave & 1) * 2;
  const int am   = lane & 15;          // A row within tile
  const int ak   = (lane >> 4) * 8;    // A k sub-chunk
  const int hcol = wg * HC;            // first global hidden column

  // B-fragment base: col = q*16 + (lane&15), k chunk = (lane>>4)*8
  const unsigned short* sWb =
      sW + ((size_t)(q * HC + (lane & 15)) * KTOT) + ((lane >> 4) * 8);

  float creg[4] = {0.f, 0.f, 0.f, 0.f};  // cell state: 4 of this WG's 64x16 tile

  for (int t = 0; t < TSTEPS; ++t) {
    const unsigned short* abuf  = ab + (size_t)(t & 1) * BATCH * DH;
    unsigned short*       anext = ab + (size_t)((t + 1) & 1) * BATCH * DH;
    const unsigned short* xt    = xb + (size_t)t * BATCH * DIN;

    // warm next step's x tile (global_prefetch_b8)
    if (t + 1 < TSTEPS) __builtin_prefetch(xt + BATCH * DIN + tid * 128, 0, 1);

    v8f acc0 = {}; v8f acc1 = {};

#pragma unroll 4
    for (int kk = 0; kk < 32; ++kk)          // x-part, K = 0..1023
      mma_k32(xt, kk * 32, sWb + kk * 32, r0, am, ak, acc0, acc1);
#pragma unroll 4
    for (int kk = 0; kk < 32; ++kk)          // a-part, K = 1024..2047
      mma_k32(abuf, kk * 32, sWb + (32 + kk) * 32, r0, am, ak, acc0, acc1);

    // ---- stage gate tiles to LDS (f32 C/D layout: M = v + 8*(lane>=16)) ----
    {
      const int n  = lane & 15;
      const int mh = (lane >> 4) * 8;
#pragma unroll
      for (int v = 0; v < 8; ++v) {
        sG[(r0 * 16 + mh + v)       * SLICE_COLS + q * HC + n] = acc0[v];
        sG[((r0 + 1) * 16 + mh + v) * SLICE_COLS + q * HC + n] = acc1[v];
      }
    }
    __syncthreads();

    // ---- elementwise: each thread owns 4 (batch,hc) cells ----
#pragma unroll
    for (int j = 0; j < 4; ++j) {
      const int flat = tid * 4 + j;
      const int row  = flat >> 4;
      const int hc   = flat & 15;
      float gf = sG[row * SLICE_COLS +  0 + hc] + bF[hcol + hc];
      float gi = sG[row * SLICE_COLS + 16 + hc] + bI[hcol + hc];
      float go = sG[row * SLICE_COLS + 32 + hc] + bO[hcol + hc];
      float gc = sG[row * SLICE_COLS + 48 + hc] + bC[hcol + hc];
      gf = sigmoidf_(gf); gi = sigmoidf_(gi); go = sigmoidf_(go);
      gc = tanhf(gc);
      const float c = gf * creg[j] + gi * gc;
      creg[j] = c;
      const float a = go * tanhf(c);
      anext[row * DH + hcol + hc] = f2bf(a);
      if (t == TSTEPS - 1) out[row * DH + hcol + hc] = a;
    }

    // ---- grid-wide barrier (release new a(t), acquire everyone's) ----
    __threadfence();
    __syncthreads();
    if (tid == 0) {
      __hip_atomic_fetch_add(bar, 1u, __ATOMIC_ACQ_REL, __HIP_MEMORY_SCOPE_AGENT);
      const unsigned target = (unsigned)(t + 1) * NWG;
      while (__hip_atomic_load(bar, __ATOMIC_ACQUIRE, __HIP_MEMORY_SCOPE_AGENT) < target)
        __builtin_amdgcn_s_sleep(2);
    }
    __syncthreads();
  }
}

// ---------------------------------------------------------------------------
extern "C" void kernel_launch(void* const* d_in, const int* in_sizes, int n_in,
                              void* d_out, int out_size, void* d_ws, size_t ws_size,
                              hipStream_t stream) {
  // setup_inputs() order: x, W_f, W_c, W_i, W_o, b_f, b_c, b_i, b_o
  const float* x   = (const float*)d_in[0];
  const float* Wf  = (const float*)d_in[1];
  const float* Wc  = (const float*)d_in[2];
  const float* Wi  = (const float*)d_in[3];
  const float* Wo  = (const float*)d_in[4];
  const float* bf_ = (const float*)d_in[5];
  const float* bc_ = (const float*)d_in[6];
  const float* bi_ = (const float*)d_in[7];
  const float* bo_ = (const float*)d_in[8];
  float* out = (float*)d_out;

  char* ws = (char*)d_ws;
  unsigned short* Wb = (unsigned short*)ws;                         // 16 MB
  unsigned short* xb = (unsigned short*)(ws + WB_ELEMS * 2);        // 64 MB
  unsigned short* ab = (unsigned short*)(ws + (WB_ELEMS + XB_ELEMS) * 2);
  unsigned*      bar = (unsigned*)(ws + (WB_ELEMS + XB_ELEMS + AB_ELEMS) * 2);

  // opt in to >64KB dynamic LDS (idempotent; harmless if already set)
  (void)hipFuncSetAttribute((const void*)k_lstm,
                            hipFuncAttributeMaxDynamicSharedMemorySize, SMEM_BYTES);

  k_init  <<<1, BLK, 0, stream>>>(ab, bar);
  k_pack_w<<<NWG, BLK, 0, stream>>>(Wf, Wi, Wo, Wc, Wb);
  {
    const size_t total = (size_t)BATCH * TSTEPS * DIN / 8;
    const int grid = (int)((total + BLK - 1) / BLK);
    k_pack_x<<<grid, BLK, 0, stream>>>(x, xb);
  }
  k_lstm<<<NWG, BLK, SMEM_BYTES, stream>>>(xb, Wb, ab, bf_, bi_, bo_, bc_, out, bar);
}
